// GCNN_RNN_Hidden_90555090469158
// MI455X (gfx1250) — compile-verified, matrix operand order checked by reading, not run on hardware
//
#include <hip/hip_runtime.h>
#include <hip/hip_bf16.h>
#include <stdint.h>

#define SEQ_LEN  64
#define NSTOCK   8000
#define NNEWS    1600
#define CIN      128
#define HID      256
#define CH       384      // CIN + HID
#define G3       768      // 3*HID
#define NEDGE    32000

typedef __bf16 bf16;
typedef __attribute__((ext_vector_type(16))) __bf16 v16bf;
typedef __attribute__((ext_vector_type(8)))  float  v8f;

// ---------------------------------------------------------------------------
// helpers
// ---------------------------------------------------------------------------
__device__ __forceinline__ void atomAddF(float* p, float v) {
    __hip_atomic_fetch_add(p, v, __ATOMIC_RELAXED, __HIP_MEMORY_SCOPE_AGENT);
}

// Load one 16x32 bf16 fragment (A or B operand) from a row-major [16][ldk]
// bf16 tile. Lane mapping per CDNA5 ISA 7.12.2:
//   lanes 0-15 : row = lane,    K 0..7 in elems 0..7,  K 16..23 in elems 8..15
//   lanes 16-31: row = lane-16, K 8..15 in elems 0..7, K 24..31 in elems 8..15
__device__ __forceinline__ v16bf load_frag(const bf16* base, int ldk, int lane) {
    const int k0 = (lane & 16) ? 8 : 0;
    const bf16* p = base + (size_t)(lane & 15) * ldk + k0;
    v16bf a;
    reinterpret_cast<float4*>(&a)[0] = *reinterpret_cast<const float4*>(p);
    reinterpret_cast<float4*>(&a)[1] = *reinterpret_cast<const float4*>(p + 16);
    return a;
}

// ---------------------------------------------------------------------------
// small elementwise kernels
// ---------------------------------------------------------------------------
__global__ void k_zero(float* __restrict__ p, int n) {
    int t = blockIdx.x * blockDim.x + threadIdx.x;
    if (t < n) p[t] = 0.0f;
}

__global__ void k_cvt_bf16(const float* __restrict__ src, bf16* __restrict__ dst, int n) {
    int t = blockIdx.x * blockDim.x + threadIdx.x;
    if (t < n) dst[t] = (bf16)src[t];
}

// broadcast a length-HID vector into rows, f32 + bf16 copies
__global__ void k_init_state(const float* __restrict__ vec, float* __restrict__ f32out,
                             bf16* __restrict__ bfout, int total) {
    int t = blockIdx.x * blockDim.x + threadIdx.x;
    if (t >= total) return;
    float v = vec[t & (HID - 1)];
    f32out[t] = v;
    bfout[t] = (bf16)v;
}

// divide segment sums by counts; emit f32 and bf16
__global__ void k_finish(const float* __restrict__ sum, const float* __restrict__ cnt,
                         float* __restrict__ outf, bf16* __restrict__ outb,
                         int rows, int cols) {
    int t = blockIdx.x * blockDim.x + threadIdx.x;
    if (t >= rows * cols) return;
    float c = cnt[t / cols];
    c = c > 1.0f ? c : 1.0f;
    float v = sum[t] / c;
    outf[t] = v;
    outb[t] = (bf16)v;
}

// ---------------------------------------------------------------------------
// edge scatters (segment-sum via f32 global atomics)
// ---------------------------------------------------------------------------
// stocks -> news : sum_news[news[e]] += x[stock[e]]  (C=128, 4 floats/thread)
__global__ void k_scatter_news(const float* __restrict__ x,
                               const int* __restrict__ news, const int* __restrict__ stock,
                               float* __restrict__ sum, float* __restrict__ cnt) {
    int t = blockIdx.x * blockDim.x + threadIdx.x;
    if (t >= NEDGE * (CIN / 4)) return;
    int e  = t >> 5;            // CIN/4 == 32 chunks per edge
    int c4 = (t & 31) * 4;
    int n = news[e], s = stock[e];
    float4 v = *reinterpret_cast<const float4*>(x + (size_t)s * CIN + c4);
    float* d = sum + (size_t)n * CIN + c4;
    atomAddF(d + 0, v.x); atomAddF(d + 1, v.y);
    atomAddF(d + 2, v.z); atomAddF(d + 3, v.w);
    if (c4 == 0) atomAddF(cnt + n, 1.0f);
}

// news -> stocks : sum_st[stock[e]] += concat(center_m, m_c)[news[e]]  (384 wide)
__global__ void k_scatter_st(const float* __restrict__ center, const float* __restrict__ mc,
                             const int* __restrict__ news, const int* __restrict__ stock,
                             float* __restrict__ sum, float* __restrict__ cnt) {
    int t = blockIdx.x * blockDim.x + threadIdx.x;
    if (t >= NEDGE * (CH / 4)) return;
    int e  = t / (CH / 4);
    int c4 = (t % (CH / 4)) * 4;
    int n = news[e], s = stock[e];
    float4 v = (c4 < CIN)
        ? *reinterpret_cast<const float4*>(center + (size_t)n * CIN + c4)
        : *reinterpret_cast<const float4*>(mc + (size_t)n * HID + (c4 - CIN));
    float* d = sum + (size_t)s * CH + c4;
    atomAddF(d + 0, v.x); atomAddF(d + 1, v.y);
    atomAddF(d + 2, v.z); atomAddF(d + 3, v.w);
    if (c4 == 0) atomAddF(cnt + s, 1.0f);
}

// ---------------------------------------------------------------------------
// fused GRU cell:  H = (1-z)*tanh(in + r*hn) + z*H
//   gi = X @ Wih^T + bih   (X: [N,KX] bf16, Wih: [768,KX] bf16)
//   gh = H @ Whh^T + bhh   (H: [N,256] bf16 stale copy)
// grid = (N/32, 2), block = (32,8): wave w -> 32 rows x 16 hidden cols
// (two 16x16 WMMA row-subtiles share every B fragment -> halved weight traffic)
// X/H row tiles streamed into double-buffered LDS with
// global_load_async_to_lds_b64 pipelined one chunk ahead (ASYNCcnt).
// The per-chunk sync is a single asm blob: wait ASYNCcnt (fill of chunk c
// done) + wait DScnt (our reads of the buffer being refilled are drained
// before we signal) + split workgroup barrier.  The blob RETURNS the LDS
// buffer-select offset, making all fragment ds_loads data-dependent on the
// barrier so the scheduler cannot hoist them above it.
// ---------------------------------------------------------------------------
template <int KX>
__global__ __launch_bounds__(256)
void k_gru(const bf16* __restrict__ Xbf,
           const bf16* __restrict__ Hbf,
           const bf16* __restrict__ Wih, const bf16* __restrict__ Whh,
           const float* __restrict__ bih, const float* __restrict__ bhh,
           float* __restrict__ Hf32, bf16* __restrict__ Hbf_out, int N) {
    __shared__ __align__(16) bf16 tile[2][32][32];

    constexpr int nci = KX >> 5;            // chunks from X
    constexpr int nct = nci + (HID >> 5);   // + chunks from Hprev

    const int lane = threadIdx.x;
    const int wave = threadIdx.y;
    const int row0 = blockIdx.x * 32;
    const int col0 = (blockIdx.y * 8 + wave) * 16;
    const int tid  = wave * 32 + lane;
    const int arow = tid >> 3;           // async-copy coords: 8 bytes/thread
    const int acol = (tid & 7) * 4;      // bf16 units
    const int klo  = (lane & 16) ? 8 : 0;
    const int wn   = lane & 15;          // B-fragment weight-row lane

    // LDS byte offsets of this thread's async destinations (low 32 bits of
    // the generic pointer are the LDS address per the aperture rules)
    const uint32_t lds_dst[2] = {
        (uint32_t)(uintptr_t)&tile[0][arow][acol],
        (uint32_t)(uintptr_t)&tile[1][arow][acol]};

    // per-thread async source base addresses (constant offsets added per chunk)
    const bf16* xsrc = Xbf + (size_t)(row0 + arow) * KX + acol;
    const bf16* hsrc = Hbf + (size_t)(row0 + arow) * HID + acol;

    // issue the async LDS fill for chunk c (8 bytes per thread, 2KB/block);
    // c is a compile-time constant after unrolling.
    auto issue = [&](int c) {
        uint64_t ga = (c < nci)
            ? (uint64_t)(uintptr_t)(xsrc + c * 32)
            : (uint64_t)(uintptr_t)(hsrc + (c - nci) * 32);
        asm volatile("global_load_async_to_lds_b64 %0, %1, off"
                     :: "v"(lds_dst[c & 1]), "v"(ga) : "memory");
    };

    const bf16* tile_flat = &tile[0][0][0];

    v8f acc[12];   // [s*6 + g] : s = row subtile, g = 0..2 gi, 3..5 gh
    #pragma unroll
    for (int i = 0; i < 12; ++i)
        #pragma unroll
        for (int j = 0; j < 8; ++j) acc[i][j] = 0.0f;

    issue(0);
    #pragma unroll
    for (int c = 0; c < nct; ++c) {
        // wait(fill c) + drain our LDS reads + barrier; returns buffer select
        uint32_t sel;
        asm volatile(
            "s_wait_asynccnt 0\n\t"
            "s_wait_dscnt 0\n\t"
            "s_barrier_signal -1\n\t"
            "s_barrier_wait -1\n\t"
            "v_mov_b32 %0, %1"
            : "=v"(sel)
            : "v"((uint32_t)((c & 1) * 32 * 32))   // bf16-element offset of buffer
            : "memory");
        if (c + 1 < nct) issue(c + 1);   // refill the other buffer in background

        const bf16* tbase = tile_flat + sel;
        v16bf a0 = load_frag(tbase,           32, lane);
        v16bf a1 = load_frag(tbase + 16 * 32, 32, lane);
        if (c < nci) {
            const int kb = c * 32;
            #pragma unroll
            for (int g = 0; g < 3; ++g) {
                const bf16* wp = Wih + (size_t)(g * HID + col0 + wn) * KX + kb + klo;
                v16bf b;
                reinterpret_cast<float4*>(&b)[0] = *reinterpret_cast<const float4*>(wp);
                reinterpret_cast<float4*>(&b)[1] = *reinterpret_cast<const float4*>(wp + 16);
                acc[g] = __builtin_amdgcn_wmma_f32_16x16x32_bf16(
                    false, a0, false, b, (short)0, acc[g], false, false);
                acc[6 + g] = __builtin_amdgcn_wmma_f32_16x16x32_bf16(
                    false, a1, false, b, (short)0, acc[6 + g], false, false);
            }
        } else {
            const int kb = (c - nci) * 32;
            #pragma unroll
            for (int g = 0; g < 3; ++g) {
                const bf16* wp = Whh + (size_t)(g * HID + col0 + wn) * HID + kb + klo;
                v16bf b;
                reinterpret_cast<float4*>(&b)[0] = *reinterpret_cast<const float4*>(wp);
                reinterpret_cast<float4*>(&b)[1] = *reinterpret_cast<const float4*>(wp + 16);
                acc[3 + g] = __builtin_amdgcn_wmma_f32_16x16x32_bf16(
                    false, a0, false, b, (short)0, acc[3 + g], false, false);
                acc[9 + g] = __builtin_amdgcn_wmma_f32_16x16x32_bf16(
                    false, a1, false, b, (short)0, acc[9 + g], false, false);
            }
        }
    }

    // ---- epilogue: biases + nonlinearities + blend ----
    // C/D layout: elem r -> M = r (+8 for lanes 16-31), N = lane&15
    const int m0  = (lane & 16) ? 8 : 0;
    const int col = col0 + wn;
    const float b_ir = bih[col],           b_hr = bhh[col];
    const float b_iz = bih[col + HID],     b_hz = bhh[col + HID];
    const float b_in = bih[col + 2 * HID], b_hn = bhh[col + 2 * HID];
    #pragma unroll
    for (int s = 0; s < 2; ++s) {
        #pragma unroll
        for (int r = 0; r < 8; ++r) {
            size_t idx = (size_t)(row0 + s * 16 + m0 + r) * HID + col;
            float ir = acc[s * 6 + 0][r] + b_ir, hr = acc[s * 6 + 3][r] + b_hr;
            float iz = acc[s * 6 + 1][r] + b_iz, hz = acc[s * 6 + 4][r] + b_hz;
            float in = acc[s * 6 + 2][r] + b_in, hn = acc[s * 6 + 5][r] + b_hn;
            float rg = 1.0f / (1.0f + __expf(-(ir + hr)));
            float zg = 1.0f / (1.0f + __expf(-(iz + hz)));
            float ng = tanhf(in + rg * hn);
            float hnew = (1.0f - zg) * ng + zg * Hf32[idx];
            Hf32[idx] = hnew;
            Hbf_out[idx] = (bf16)hnew;
        }
    }
}

// ---------------------------------------------------------------------------
// host
// ---------------------------------------------------------------------------
static inline int cdiv(long a, int b) { return (int)((a + b - 1) / b); }

extern "C" void kernel_launch(void* const* d_in, const int* in_sizes, int n_in,
                              void* d_out, int out_size, void* d_ws, size_t ws_size,
                              hipStream_t stream) {
    (void)in_sizes; (void)n_in; (void)out_size; (void)ws_size;
    const float* inputs   = (const float*)d_in[0];
    const int*   edges    = (const int*)d_in[1];
    const float* h_start  = (const float*)d_in[2];
    const float* hi_start = (const float*)d_in[3];
    const float* mc_start = (const float*)d_in[4];
    const float* Wih_c  = (const float*)d_in[5];
    const float* Whh_c  = (const float*)d_in[6];
    const float* bih_c  = (const float*)d_in[7];
    const float* bhh_c  = (const float*)d_in[8];
    const float* Wih_lh = (const float*)d_in[9];
    const float* Whh_lh = (const float*)d_in[10];
    const float* bih_lh = (const float*)d_in[11];
    const float* bhh_lh = (const float*)d_in[12];
    const float* Wih_li = (const float*)d_in[13];
    const float* Whh_li = (const float*)d_in[14];
    const float* bih_li = (const float*)d_in[15];
    const float* bhh_li = (const float*)d_in[16];

    // outputs: m [8000x384] | h [8000x256] | h_input [8000x256]
    float* m_out  = (float*)d_out;
    float* h_f32  = m_out + (size_t)NSTOCK * CH;
    float* hi_f32 = h_f32 + (size_t)NSTOCK * HID;

    // ---- carve workspace ----
    char* ws = (char*)d_ws;
    auto alloc = [&](size_t bytes) -> char* {
        char* p = ws; ws += (bytes + 255) & ~(size_t)255; return p;
    };
    bf16* Wih_c_b  = (bf16*)alloc((size_t)G3 * CIN * 2);
    bf16* Whh_c_b  = (bf16*)alloc((size_t)G3 * HID * 2);
    bf16* Wih_lh_b = (bf16*)alloc((size_t)G3 * CH  * 2);
    bf16* Whh_lh_b = (bf16*)alloc((size_t)G3 * HID * 2);
    bf16* Wih_li_b = (bf16*)alloc((size_t)G3 * CIN * 2);
    bf16* Whh_li_b = (bf16*)alloc((size_t)G3 * HID * 2);
    bf16* h_bf[2]  = {(bf16*)alloc((size_t)NSTOCK * HID * 2), (bf16*)alloc((size_t)NSTOCK * HID * 2)};
    bf16* hi_bf[2] = {(bf16*)alloc((size_t)NSTOCK * HID * 2), (bf16*)alloc((size_t)NSTOCK * HID * 2)};
    bf16* mc_bf[2] = {(bf16*)alloc((size_t)NNEWS * HID * 2), (bf16*)alloc((size_t)NNEWS * HID * 2)};
    bf16* x_bf      = (bf16*)alloc((size_t)NSTOCK * CIN * 2);
    bf16* center_bf = (bf16*)alloc((size_t)NNEWS * CIN * 2);
    bf16* m_bf      = (bf16*)alloc((size_t)NSTOCK * CH * 2);
    float* mc_f32     = (float*)alloc((size_t)NNEWS * HID * 4);
    float* center_f32 = (float*)alloc((size_t)NNEWS * CIN * 4);
    // contiguous zero block: sum_news | cnt_news | sum_st | cnt_st
    const long ZN = (long)NNEWS * CIN + NNEWS + (long)NSTOCK * CH + NSTOCK;
    float* sum_news = (float*)alloc((size_t)ZN * 4);
    float* cnt_news = sum_news + (size_t)NNEWS * CIN;
    float* sum_st   = cnt_news + NNEWS;
    float* cnt_st   = sum_st + (size_t)NSTOCK * CH;

    const int B = 256;
    // ---- one-time per call: weight conversion + state init ----
    k_cvt_bf16<<<cdiv((long)G3 * CIN, B), B, 0, stream>>>(Wih_c,  Wih_c_b,  G3 * CIN);
    k_cvt_bf16<<<cdiv((long)G3 * HID, B), B, 0, stream>>>(Whh_c,  Whh_c_b,  G3 * HID);
    k_cvt_bf16<<<cdiv((long)G3 * CH,  B), B, 0, stream>>>(Wih_lh, Wih_lh_b, G3 * CH);
    k_cvt_bf16<<<cdiv((long)G3 * HID, B), B, 0, stream>>>(Whh_lh, Whh_lh_b, G3 * HID);
    k_cvt_bf16<<<cdiv((long)G3 * CIN, B), B, 0, stream>>>(Wih_li, Wih_li_b, G3 * CIN);
    k_cvt_bf16<<<cdiv((long)G3 * HID, B), B, 0, stream>>>(Whh_li, Whh_li_b, G3 * HID);
    k_init_state<<<cdiv((long)NSTOCK * HID, B), B, 0, stream>>>(h_start,  h_f32,  h_bf[0],  NSTOCK * HID);
    k_init_state<<<cdiv((long)NSTOCK * HID, B), B, 0, stream>>>(hi_start, hi_f32, hi_bf[0], NSTOCK * HID);
    k_init_state<<<cdiv((long)NNEWS  * HID, B), B, 0, stream>>>(mc_start, mc_f32, mc_bf[0], NNEWS * HID);

    const dim3 gruBlk(32, 8);
    const dim3 gruGridS(NSTOCK / 32, 2);
    const dim3 gruGridN(NNEWS / 32, 2);

    for (int t = 0; t < SEQ_LEN; ++t) {
        const float* x_t  = inputs + (size_t)t * NSTOCK * CIN;
        const int* news   = edges + (size_t)t * 2 * NEDGE;
        const int* stock  = news + NEDGE;
        const int cur = t & 1, nxt = cur ^ 1;

        k_zero<<<cdiv(ZN, B), B, 0, stream>>>(sum_news, (int)ZN);
        k_scatter_news<<<cdiv((long)NEDGE * (CIN / 4), B), B, 0, stream>>>(
            x_t, news, stock, sum_news, cnt_news);
        k_finish<<<cdiv((long)NNEWS * CIN, B), B, 0, stream>>>(
            sum_news, cnt_news, center_f32, center_bf, NNEWS, CIN);
        k_gru<CIN><<<gruGridN, gruBlk, 0, stream>>>(
            center_bf, mc_bf[cur], Wih_c_b, Whh_c_b, bih_c, bhh_c,
            mc_f32, mc_bf[nxt], NNEWS);
        k_scatter_st<<<cdiv((long)NEDGE * (CH / 4), B), B, 0, stream>>>(
            center_f32, mc_f32, news, stock, sum_st, cnt_st);
        k_finish<<<cdiv((long)NSTOCK * CH, B), B, 0, stream>>>(
            sum_st, cnt_st, m_out, m_bf, NSTOCK, CH);
        k_gru<CH><<<gruGridS, gruBlk, 0, stream>>>(
            m_bf, h_bf[cur], Wih_lh_b, Whh_lh_b, bih_lh, bhh_lh,
            h_f32, h_bf[nxt], NSTOCK);
        k_cvt_bf16<<<cdiv((long)NSTOCK * CIN, B), B, 0, stream>>>(x_t, x_bf, NSTOCK * CIN);
        k_gru<CIN><<<gruGridS, gruBlk, 0, stream>>>(
            x_bf, hi_bf[cur], Wih_li_b, Whh_li_b, bih_li, bhh_li,
            hi_f32, hi_bf[nxt], NSTOCK);
    }
}
